// ModulatedDeformConv_78151224918576
// MI455X (gfx1250) — compile-verified, matrix operand order checked by reading, not run on hardware
//
#include <hip/hip_runtime.h>
#include <math.h>

// ---------------- problem constants (from reference) ----------------
#define NB   8
#define CI   64
#define HH   96
#define WW   96
#define HWW  (HH*WW)          // 9216
#define CHW  (CI*HWW)         // 589824
#define CO_  64
#define KT   576              // C*KK, K ordered as k = kk*64 + c
#define MTOT (NB*HWW)         // 73728 output pixels

typedef __attribute__((ext_vector_type(16))) _Float16 v16h;
typedef __attribute__((ext_vector_type(8)))  _Float16 v8h;
typedef __attribute__((ext_vector_type(8)))  float    v8f;

// workspace layout (bytes):
//   [0, 73728)          Bmain f16 [64][576]   (co-major, K = kk*64+c)
//   [73728, 110592)     Boff  f16 [32][576]   (rows 27..31 zero-padded)
//   [110592, 8073216)   offmask float [8][27][9216]  (ch 0..17 raw, 18..26 sigmoid)
#define WS_BOFF   73728
#define WS_OFFM   110592

// LDS A tile: 64 pixels x 64 K halves, row stride 72 halves (144 B):
//  - 16B-aligned rows/fragments, bank-conflict-free ds_read_b128 (gcd(36,64) step ok)
#define ASTRIDE 72

// ---------------- weight repack: fp32 OIHW -> f16 K-major ----------------
__global__ __launch_bounds__(256)
void prep_kernel(const float* __restrict__ weight, const float* __restrict__ w_off,
                 _Float16* __restrict__ Bmain, _Float16* __restrict__ Boff) {
  int i = blockIdx.x * 256 + threadIdx.x;            // 144*256 = 36864 = 64*576
  {
    int co = i / KT, r = i % KT;
    int kk = r >> 6, c = r & 63;                     // k = kk*64 + c
    Bmain[i] = (_Float16)weight[co * KT + c * 9 + kk];
  }
  if (i < 32 * KT) {
    int o = i / KT, r = i % KT;
    int kk = r >> 6, c = r & 63;
    Boff[i] = (o < 27) ? (_Float16)w_off[o * KT + c * 9 + kk] : (_Float16)0.f;
  }
}

// B fragment: two 16B chunks (halves 0..7 = K kbase..+7, 8..15 = K kbase+16..+23)
union Frag16 { v16h v; uint4 q[2]; };

__device__ __forceinline__ v16h load_bfrag(const _Float16* __restrict__ B,
                                           int row, int kbase) {
  Frag16 f;
  const char* p = (const char*)(B + row * KT + kbase);
  f.q[0] = *(const uint4*)p;
  f.q[1] = *(const uint4*)(p + 32);
  return f.v;
}

// ---------------- kernel 1: offset conv (im2col GEMM, 27 outch) ----------------
__global__ __launch_bounds__(128)
void offset_conv_kernel(const float* __restrict__ x, const _Float16* __restrict__ Boff,
                        const float* __restrict__ b_off, float* __restrict__ offmask) {
  __shared__ _Float16 A[64 * ASTRIDE];   // 64 pixels x 64 K (9216 B)

  const int tid  = threadIdx.x;
  const int lane = tid & 31;             // wave32
  const int wave = tid >> 5;             // 4 waves
  const int mb   = blockIdx.x * 64;

  // staging identity: thread -> (pixel, 32-channel band)
  const int ps    = tid & 63;
  const int cband = (tid >> 6) << 5;     // 0 or 32
  const int p_g   = mb + ps;
  const int n_s   = p_g / HWW;
  const int rem_s = p_g % HWW;
  const int h_s   = rem_s / WW, w_s = rem_s % WW;

  const int nidx  = lane & 15;
  const int khalf = (lane >> 4) << 3;
  const _Float16* arow_mma = &A[(wave * 16 + nidx) * ASTRIDE];

  v8f acc0 = {}, acc1 = {};

  for (int t = 0; t < 9; ++t) {          // one kernel point (K=64) per iteration
    const int ky = t / 3, kx = t % 3;
    __syncthreads();
    // ---- stage im2col A tile, branchless + packed b128 LDS stores ----
    {
      const int y = h_s + ky - 1, xx = w_s + kx - 1;
      const float wgt = ((y >= 0) & (y < HH) & (xx >= 0) & (xx < WW)) ? 1.f : 0.f;
      const int yc = min(max(y, 0), HH - 1), xc = min(max(xx, 0), WW - 1);
      const float* px = x + (size_t)n_s * CHW + (size_t)cband * HWW + yc * WW + xc;
      _Float16* arow = &A[ps * ASTRIDE + cband];
#pragma unroll
      for (int g = 0; g < 4; ++g) {      // 4 groups of 8 channels
        v8h h;
#pragma unroll
        for (int j = 0; j < 8; ++j)
          h[j] = (_Float16)(wgt * px[(size_t)(g * 8 + j) * HWW]);
        *(v8h*)(arow + g * 8) = h;       // ds_store_b128
      }
    }
    __syncthreads();
    // ---- WMMA: 16 pixels x 32 outch per wave, K=64 ----
    Frag16 af0, af1;
    af0.q[0] = *(const uint4*)(arow_mma + khalf);
    af0.q[1] = *(const uint4*)(arow_mma + khalf + 16);
    af1.q[0] = *(const uint4*)(arow_mma + 32 + khalf);
    af1.q[1] = *(const uint4*)(arow_mma + 32 + khalf + 16);
    const int k0 = t * 64;
    v16h b00 = load_bfrag(Boff, nidx,      k0 + khalf);
    v16h b01 = load_bfrag(Boff, nidx,      k0 + 32 + khalf);
    v16h b10 = load_bfrag(Boff, 16 + nidx, k0 + khalf);
    v16h b11 = load_bfrag(Boff, 16 + nidx, k0 + 32 + khalf);
    acc0 = __builtin_amdgcn_wmma_f32_16x16x32_f16(false, af0.v, false, b00, (short)0, acc0, false, false);
    acc0 = __builtin_amdgcn_wmma_f32_16x16x32_f16(false, af1.v, false, b01, (short)0, acc0, false, false);
    acc1 = __builtin_amdgcn_wmma_f32_16x16x32_f16(false, af0.v, false, b10, (short)0, acc1, false, false);
    acc1 = __builtin_amdgcn_wmma_f32_16x16x32_f16(false, af1.v, false, b11, (short)0, acc1, false, false);
  }

  // ---- epilogue: bias, sigmoid on mask channels, store [N][27][H][W] ----
  const int lg = lane >> 4;
  const float bs0 = b_off[nidx];
  const int oc1 = 16 + nidx;
  const float bs1 = (oc1 < 27) ? b_off[oc1] : 0.f;
#pragma unroll
  for (int v = 0; v < 8; ++v) {
    const int m = v + 8 * lg;
    const int p = mb + wave * 16 + m;
    const int n = p / HWW, rem = p % HWW;
    offmask[((size_t)n * 27 + nidx) * HWW + rem] = acc0[v] + bs0;   // ch 0..15: raw
    if (oc1 < 27) {
      float z = acc1[v] + bs1;
      if (oc1 >= 18) z = 1.f / (1.f + __expf(-z));                  // mask: sigmoid
      offmask[((size_t)n * 27 + oc1) * HWW + rem] = z;
    }
  }
}

// ---------------- kernel 2: deformable sampling + main GEMM ----------------
__global__ __launch_bounds__(128)
void deform_gemm_kernel(const float* __restrict__ x, const _Float16* __restrict__ Bmain,
                        const float* __restrict__ offmask, const float* __restrict__ bias,
                        float* __restrict__ out) {
  __shared__ _Float16 A[64 * ASTRIDE];
  __shared__ float SPw[4][64];   // per-pixel corner weights (mask & validity folded in)
  __shared__ int   SPo[4][64];   // per-pixel clamped corner offsets y*W+x

  const int tid  = threadIdx.x;
  const int lane = tid & 31;
  const int wave = tid >> 5;
  const int mb   = blockIdx.x * 64;

  const int ps    = tid & 63;
  const int cband = (tid >> 6) << 5;     // 0 or 32
  const int p_g   = mb + ps;
  const int n_s   = p_g / HWW;

  const int nidx  = lane & 15;
  const int khalf = (lane >> 4) << 3;
  const _Float16* arow_mma = &A[(wave * 16 + nidx) * ASTRIDE];

  v8f acc[4] = {};

  for (int t = 0; t < 9; ++t) {          // one kernel point (K=64) per iteration
    __syncthreads();
    if (tid < 64) {
      // ---- bilinear sampling parameters for kernel point t, 64 pixels ----
      const int p = mb + tid;
      const int n = p / HWW, rem = p % HWW;
      const int h = rem / WW, w = rem % WW;
      const size_t ob = (size_t)n * 27 * HWW + rem;
      const float dy = offmask[ob + (size_t)(2 * t) * HWW];
      const float dx = offmask[ob + (size_t)(2 * t + 1) * HWW];
      const float mk = offmask[ob + (size_t)(18 + t) * HWW];
      const int ky = t / 3, kx = t % 3;
      const float py = dy + (float)(ky + h - 1);
      const float px = dx + (float)(kx + w - 1);
      const float y0f = floorf(py), x0f = floorf(px);
      const float ly = py - y0f, lx = px - x0f;
      const int y0 = (int)y0f, x0 = (int)x0f, y1 = y0 + 1, x1 = x0 + 1;
      const float vy0 = (y0 >= 0 && y0 < HH) ? 1.f : 0.f;
      const float vy1 = (y1 >= 0 && y1 < HH) ? 1.f : 0.f;
      const float vx0 = (x0 >= 0 && x0 < WW) ? 1.f : 0.f;
      const float vx1 = (x1 >= 0 && x1 < WW) ? 1.f : 0.f;
      SPw[0][tid] = (1.f - ly) * (1.f - lx) * mk * vy0 * vx0;
      SPw[1][tid] = (1.f - ly) * lx         * mk * vy0 * vx1;
      SPw[2][tid] = ly * (1.f - lx)         * mk * vy1 * vx0;
      SPw[3][tid] = ly * lx                 * mk * vy1 * vx1;
      const int y0c = min(max(y0, 0), HH - 1), y1c = min(max(y1, 0), HH - 1);
      const int x0c = min(max(x0, 0), WW - 1), x1c = min(max(x1, 0), WW - 1);
      SPo[0][tid] = y0c * WW + x0c;
      SPo[1][tid] = y0c * WW + x1c;
      SPo[2][tid] = y1c * WW + x0c;
      SPo[3][tid] = y1c * WW + x1c;
    }
    __syncthreads();
    // ---- gather + modulate -> f16 A tile (lanes = pixels -> coalesced) ----
    {
      const float w00 = SPw[0][ps], w01 = SPw[1][ps], w10 = SPw[2][ps], w11 = SPw[3][ps];
      const int   o00 = SPo[0][ps], o01 = SPo[1][ps], o10 = SPo[2][ps], o11 = SPo[3][ps];
      const float* bp = x + (size_t)n_s * CHW + (size_t)cband * HWW;
      _Float16* arow = &A[ps * ASTRIDE + cband];
#pragma unroll
      for (int g = 0; g < 4; ++g) {      // 4 groups of 8 channels
        v8h hgrp;
#pragma unroll
        for (int j = 0; j < 8; ++j) {
          const float* pc = bp + (size_t)(g * 8 + j) * HWW;
          hgrp[j] = (_Float16)(w00 * pc[o00] + w01 * pc[o01] + w10 * pc[o10] + w11 * pc[o11]);
        }
        *(v8h*)(arow + g * 8) = hgrp;    // ds_store_b128
      }
    }
    __syncthreads();
    // ---- WMMA: 16 pixels x 64 co per wave, K=64 (8 back-to-back WMMAs) ----
    Frag16 af0, af1;
    af0.q[0] = *(const uint4*)(arow_mma + khalf);
    af0.q[1] = *(const uint4*)(arow_mma + khalf + 16);
    af1.q[0] = *(const uint4*)(arow_mma + 32 + khalf);
    af1.q[1] = *(const uint4*)(arow_mma + 32 + khalf + 16);
    const int k0 = t * 64;
#pragma unroll
    for (int nt = 0; nt < 4; ++nt) {
      v16h bf0 = load_bfrag(Bmain, nt * 16 + nidx, k0 + khalf);
      v16h bf1 = load_bfrag(Bmain, nt * 16 + nidx, k0 + 32 + khalf);
      acc[nt] = __builtin_amdgcn_wmma_f32_16x16x32_f16(false, af0.v, false, bf0, (short)0, acc[nt], false, false);
      acc[nt] = __builtin_amdgcn_wmma_f32_16x16x32_f16(false, af1.v, false, bf1, (short)0, acc[nt], false, false);
    }
  }

  // ---- epilogue: bias, store [N][64][H][W] fp32 ----
  const int lg = lane >> 4;
#pragma unroll
  for (int nt = 0; nt < 4; ++nt) {
    const int co = nt * 16 + nidx;
    const float bs = bias[co];
#pragma unroll
    for (int v = 0; v < 8; ++v) {
      const int m = v + 8 * lg;
      const int p = mb + wave * 16 + m;
      const int n = p / HWW, rem = p % HWW;
      out[((size_t)n * CO_ + co) * HWW + rem] = acc[nt][v] + bs;
    }
  }
}

// ---------------- launcher ----------------
extern "C" void kernel_launch(void* const* d_in, const int* in_sizes, int n_in,
                              void* d_out, int out_size, void* d_ws, size_t ws_size,
                              hipStream_t stream) {
  const float* x      = (const float*)d_in[0];
  const float* w_off  = (const float*)d_in[1];
  const float* b_off  = (const float*)d_in[2];
  const float* weight = (const float*)d_in[3];
  const float* bias   = (const float*)d_in[4];
  float* out = (float*)d_out;

  char* ws = (char*)d_ws;
  _Float16* Bmain  = (_Float16*)ws;
  _Float16* Boff   = (_Float16*)(ws + WS_BOFF);
  float*    offmsk = (float*)(ws + WS_OFFM);

  prep_kernel<<<144, 256, 0, stream>>>(weight, w_off, Bmain, Boff);
  offset_conv_kernel<<<MTOT / 64, 128, 0, stream>>>(x, Boff, b_off, offmsk);
  deform_gemm_kernel<<<MTOT / 64, 128, 0, stream>>>(x, Bmain, offmsk, bias, out);
}
